// AttentionBlock_19335942766924
// MI455X (gfx1250) — matrix operand order repaired and checked by reading, not compiled, not run on hardware
//
#include <hip/hip_runtime.h>

// ============================================================================
// AttentionBlock for MI455X (gfx1250, wave32, WMMA).
// Compute-bound (~86 GFLOP vs ~35MB HBM) -> all big matmuls use
// v_wmma_f32_16x16x32_bf16 (f32 accumulate), softmax/LN stats in f32.
// Round 3: attention reuses K/V register fragments across TWO query tiles
// (halves L2 traffic 4GB->2GB, the binding constraint), parallel LN with
// coalesced bf16 write-out, batched B-fragment ds_loads in the GEMMs.
// ============================================================================

#define BATCH 2
#define CH    512
#define NTOK  4096   // 64*64
#define DQKV  1536   // 3*CH
#define LDW   (CH + 8)   // padded LDS row stride (elements)

typedef __bf16 bf16;
typedef bf16  v16bf __attribute__((ext_vector_type(16)));
typedef bf16  v8bf  __attribute__((ext_vector_type(8)));
typedef float v8f   __attribute__((ext_vector_type(8)));

// ---------------------------------------------------------------- helpers ---
__device__ __forceinline__ unsigned short f2bf(float f) {
  unsigned u = __builtin_bit_cast(unsigned, f);
  u += 0x7FFFu + ((u >> 16) & 1u);           // round-to-nearest-even
  return (unsigned short)(u >> 16);
}

__device__ __forceinline__ v8f wmma_bf16(v16bf a, v16bf b, v8f c) {
  return __builtin_amdgcn_wmma_f32_16x16x32_bf16(false, a, false, b,
                                                 (short)0, c, false, false);
}

// A-fragment (16x32, bf16) from row-major src (leading dim ld elements).
__device__ __forceinline__ v16bf load_a_frag(const unsigned short* p, int ld,
                                             int lane) {
  int row = lane & 15;
  int kb  = (lane >> 4) * 16;
  const bf16* q = (const bf16*)p;
  v8bf lo = *(const v8bf*)(q + (size_t)row * ld + kb);
  v8bf hi = *(const v8bf*)(q + (size_t)row * ld + kb + 8);
  return __builtin_shufflevector(lo, hi, 0, 1, 2, 3, 4, 5, 6, 7,
                                 8, 9, 10, 11, 12, 13, 14, 15);
}

// B-fragment (32x16): B[k][n] = W[n][k], W row-major; one 32B read per lane.
__device__ __forceinline__ v16bf load_b_frag(const unsigned short* p, int ld,
                                             int lane) {
  int col = lane & 15;
  int kb  = (lane >> 4) * 16;
  const bf16* q = (const bf16*)p;
  return *(const v16bf*)(q + (size_t)col * ld + kb);
}

// Same but via two 16B reads (padded LDS rows are only 16B aligned).
__device__ __forceinline__ v16bf load_b_frag16(const unsigned short* p, int ld,
                                               int lane) {
  int col = lane & 15;
  int kb  = (lane >> 4) * 16;
  const bf16* q = (const bf16*)p;
  v8bf lo = *(const v8bf*)(q + (size_t)col * ld + kb);
  v8bf hi = *(const v8bf*)(q + (size_t)col * ld + kb + 8);
  return __builtin_shufflevector(lo, hi, 0, 1, 2, 3, 4, 5, 6, 7,
                                 8, 9, 10, 11, 12, 13, 14, 15);
}

// ------------------------------------------------------------ kernel 0: cvt -
__global__ void cvt_f32_bf16(const float* __restrict__ src,
                             unsigned short* __restrict__ dst, int n) {
  int i = blockIdx.x * blockDim.x + threadIdx.x;
  if (i < n) dst[i] = f2bf(src[i]);
}

// --------------------------------------------------- kernel 1: layernorm ----
// Block = 256 threads, 64 rows; C split 4-way with LDS reduce; normalized
// tile staged in padded LDS, then coalesced vectorized bf16 write-out.
__global__ void ln_kernel(const float* __restrict__ x,
                          const float* __restrict__ norm_w,
                          unsigned short* __restrict__ xn) {
  __shared__ float s_s[4][64], s_s2[4][64];
  __shared__ float s_mu[64], s_r[64];
  __shared__ __align__(16) unsigned short s_x[64 * LDW];   // ~65 KB

  const int rows0 = blockIdx.x * 64;          // global row = b*NTOK + n
  const int b  = rows0 / NTOK;
  const int n0 = rows0 % NTOK;                // 64-aligned, stays in batch
  const int nl = threadIdx.x & 63, cg = threadIdx.x >> 6;
  const float* xb = x + (size_t)b * CH * NTOK + n0;

  float s = 0.f, ss = 0.f;
  for (int c = cg * 128; c < cg * 128 + 128; ++c) {
    float v = xb[(size_t)c * NTOK + nl];
    s += v; ss += v * v;
  }
  s_s[cg][nl] = s; s_s2[cg][nl] = ss;
  __syncthreads();
  if (threadIdx.x < 64) {
    float st = 0.f, sst = 0.f;
#pragma unroll
    for (int g = 0; g < 4; ++g) { st += s_s[g][threadIdx.x];
                                  sst += s_s2[g][threadIdx.x]; }
    const float inv = 1.0f / CH;
    float mu = st * inv;
    float var = sst * inv - mu * mu;
    s_mu[threadIdx.x] = mu;
    s_r[threadIdx.x]  = rsqrtf(var + 1e-5f);
  }
  __syncthreads();
  {
    float mu = s_mu[nl], r = s_r[nl];
    for (int c = cg * 128; c < cg * 128 + 128; ++c) {
      float v = (xb[(size_t)c * NTOK + nl] - mu) * r * norm_w[c];
      s_x[nl * LDW + c] = f2bf(v);
    }
  }
  __syncthreads();
  for (int i = threadIdx.x; i < 64 * (CH / 8); i += 256) {
    int rr = i >> 6, cc = (i & 63) * 8;
    *(v8bf*)((bf16*)xn + (size_t)(rows0 + rr) * CH + cc) =
        *(const v8bf*)((const bf16*)s_x + rr * LDW + cc);
  }
}

// --------------------------------------------------- kernel 2: QKV GEMM -----
__global__ void qkv_gemm(const unsigned short* __restrict__ xn,
                         const unsigned short* __restrict__ wq,
                         const float* __restrict__ bias,
                         unsigned short* __restrict__ qb,
                         unsigned short* __restrict__ kb_,
                         unsigned short* __restrict__ vt) {
  __shared__ __align__(16) unsigned short s_w[64 * LDW];   // ~65 KB
  const int lane = threadIdx.x & 31;
  const int wave = threadIdx.x >> 5;
  const int m0 = (blockIdx.x * 8 + wave) * 16;   // token rows
  const int d0 = blockIdx.y * 64;                // qkv output cols
  const int b  = blockIdx.z;
  const float SCALE = 0.04419417382415922f;      // 512^-0.5

  for (int i = threadIdx.x; i < 64 * (CH / 8); i += 256) {
    int r = i >> 6, ck = (i & 63) * 8;
    *(v8bf*)(s_w + r * LDW + ck) =
        *(const v8bf*)((const bf16*)wq + (size_t)(d0 + r) * CH + ck);
  }
  __syncthreads();

  const unsigned short* A = xn + ((size_t)b * NTOK + m0) * CH;
  v8f acc[4] = {v8f{0}, v8f{0}, v8f{0}, v8f{0}};
  v16bf a_cur = load_a_frag(A, CH, lane);
  for (int k = 0; k < CH; k += 32) {
    v16bf a_nxt = load_a_frag(A + ((k + 32) & (CH - 1)), CH, lane);
    v16bf bfr[4];
#pragma unroll
    for (int t = 0; t < 4; ++t)
      bfr[t] = load_b_frag16(s_w + (size_t)(t * 16) * LDW + k, LDW, lane);
#pragma unroll
    for (int t = 0; t < 4; ++t) acc[t] = wmma_bf16(a_cur, bfr[t], acc[t]);
    a_cur = a_nxt;
  }
  const int col = lane & 15, hi = lane >> 4;
#pragma unroll
  for (int t = 0; t < 4; ++t) {
    int d = d0 + t * 16 + col;
    float bv = bias[d];
    if (d < CH) {                       // Q, pre-scaled
#pragma unroll
      for (int v = 0; v < 8; ++v) {
        int m = m0 + v + 8 * hi;
        qb[((size_t)b * NTOK + m) * CH + d] = f2bf((acc[t][v] + bv) * SCALE);
      }
    } else if (d < 2 * CH) {            // K
#pragma unroll
      for (int v = 0; v < 8; ++v) {
        int m = m0 + v + 8 * hi;
        kb_[((size_t)b * NTOK + m) * CH + (d - CH)] = f2bf(acc[t][v] + bv);
      }
    } else {                            // V transposed: vt[b][c][n]
#pragma unroll
      for (int v = 0; v < 8; ++v) {
        int m = m0 + v + 8 * hi;
        vt[((size_t)b * CH + (d - 2 * CH)) * NTOK + m] = f2bf(acc[t][v] + bv);
      }
    }
  }
}

// --------------------------------------------------- kernel 3: attention ----
// Block = 8 waves, 32 query rows (2 tiles). Wave w owns C-slice [64w,64w+64).
// K/V fragments are loaded once per 32-key block and reused (in registers)
// for BOTH query tiles -> halves K/V L2 traffic per query.
__global__ void attn_kernel(const unsigned short* __restrict__ qb,
                            const unsigned short* __restrict__ kb_,
                            const unsigned short* __restrict__ vt,
                            unsigned short* __restrict__ ob) {
  __shared__ float s_part[2][8][16][32];             // 32 KB
  __shared__ float s_sum[2][16][32];                 //  4 KB
  __shared__ __align__(32) unsigned short s_p[2][16][32];
  __shared__ float s_m[32], s_l[32], s_alpha[32];

  const int lane = threadIdx.x & 31;
  const int wave = threadIdx.x >> 5;
  const int q0 = blockIdx.x * 32;
  const int b  = blockIdx.y;
  const int cs = wave * 64;                          // this wave's C slice
  const int col = lane & 15, hi = lane >> 4;

  if (threadIdx.x < 32) { s_m[threadIdx.x] = -1e30f; s_l[threadIdx.x] = 0.f; }

  // Q fragments for both query tiles, this wave's C-slice (resident)
  v16bf aq[2][2];
#pragma unroll
  for (int qt = 0; qt < 2; ++qt) {
    const unsigned short* qp = qb + ((size_t)b * NTOK + q0 + qt * 16) * CH + cs;
    aq[qt][0] = load_a_frag(qp, CH, lane);
    aq[qt][1] = load_a_frag(qp + 32, CH, lane);
  }
  v8f o[2][4] = {};
  __syncthreads();

  for (int j0 = 0; j0 < NTOK; j0 += 32) {
    // ---- V fragments early (shared by both query tiles)
    v16bf bv[4];
#pragma unroll
    for (int t = 0; t < 4; ++t)
      bv[t] = load_b_frag(vt + ((size_t)b * CH + cs + t * 16) * NTOK + j0,
                          NTOK, lane);
    {
      int jn = (j0 + 32) & (NTOK - 1);
      __builtin_prefetch(kb_ + ((size_t)b * NTOK + jn) * CH + cs, 0, 3);
      __builtin_prefetch(vt + ((size_t)b * CH + cs) * NTOK + jn, 0, 3);
    }
    // ---- K fragments once; S partials for both query tiles
    const unsigned short* kp = kb_ + ((size_t)b * NTOK + j0) * CH + cs;
    v16bf bk0 = load_b_frag(kp, CH, lane);
    v16bf bk1 = load_b_frag(kp + (size_t)16 * CH, CH, lane);
    v16bf bk2 = load_b_frag(kp + 32, CH, lane);
    v16bf bk3 = load_b_frag(kp + (size_t)16 * CH + 32, CH, lane);
#pragma unroll
    for (int qt = 0; qt < 2; ++qt) {
      v8f s0 = v8f{0}, s1 = v8f{0};
      s0 = wmma_bf16(aq[qt][0], bk0, s0);
      s1 = wmma_bf16(aq[qt][0], bk1, s1);
      s0 = wmma_bf16(aq[qt][1], bk2, s0);
      s1 = wmma_bf16(aq[qt][1], bk3, s1);
#pragma unroll
      for (int v = 0; v < 8; ++v) {
        s_part[qt][wave][v + 8 * hi][col]      = s0[v];
        s_part[qt][wave][v + 8 * hi][16 + col] = s1[v];
      }
    }
    __syncthreads();

    // ---- reduce the 8 partials (256 threads x 4 elements)
    for (int e = threadIdx.x; e < 1024; e += 256) {
      int qt = e >> 9, m = (e >> 5) & 15, j = e & 31;
      float a = 0.f;
#pragma unroll
      for (int w = 0; w < 8; ++w) a += s_part[qt][w][m][j];
      s_sum[qt][m][j] = a;
    }
    __syncthreads();

    // ---- online softmax: 16 threads per row, width-16 shuffle butterflies
#pragma unroll
    for (int qt = 0; qt < 2; ++qt) {
      int row = threadIdx.x >> 4, sub = threadIdx.x & 15;
      float e0 = s_sum[qt][row][sub], e1 = s_sum[qt][row][sub + 16];
      float mx = fmaxf(e0, e1);
#pragma unroll
      for (int off = 8; off; off >>= 1) mx = fmaxf(mx, __shfl_xor(mx, off, 16));
      float mold = s_m[qt * 16 + row];
      float mnew = fmaxf(mold, mx);
      float p0 = __expf(e0 - mnew), p1 = __expf(e1 - mnew);
      float ps = p0 + p1;
#pragma unroll
      for (int off = 8; off; off >>= 1) ps += __shfl_xor(ps, off, 16);
      s_p[qt][row][sub]      = f2bf(p0);
      s_p[qt][row][sub + 16] = f2bf(p1);
      if (sub == 0) {
        float alpha = __expf(mold - mnew);
        s_alpha[qt * 16 + row] = alpha;
        s_l[qt * 16 + row] = s_l[qt * 16 + row] * alpha + ps;
        s_m[qt * 16 + row] = mnew;
      }
    }
    __syncthreads();

    // ---- O = O*alpha + P @ V[:, slice] (bv reused across both tiles)
#pragma unroll
    for (int qt = 0; qt < 2; ++qt) {
      float al[8];
#pragma unroll
      for (int v = 0; v < 8; ++v) al[v] = s_alpha[qt * 16 + v + 8 * hi];
      v16bf ap = load_a_frag(&s_p[qt][0][0], 32, lane);
#pragma unroll
      for (int t = 0; t < 4; ++t) {
#pragma unroll
        for (int v = 0; v < 8; ++v) o[qt][t][v] *= al[v];
        o[qt][t] = wmma_bf16(ap, bv[t], o[qt][t]);
      }
    }
    __syncthreads();   // s_p / s_sum / stats free for next iteration
  }

  // ---- normalize by l, store O bf16 [b][n][c]
#pragma unroll
  for (int qt = 0; qt < 2; ++qt) {
    float linv[8];
#pragma unroll
    for (int v = 0; v < 8; ++v) linv[v] = 1.0f / s_l[qt * 16 + v + 8 * hi];
#pragma unroll
    for (int t = 0; t < 4; ++t)
#pragma unroll
      for (int v = 0; v < 8; ++v) {
        int n = q0 + qt * 16 + v + 8 * hi;
        ob[((size_t)b * NTOK + n) * CH + cs + t * 16 + col] =
            f2bf(o[qt][t][v] * linv[v]);
      }
  }
}

// --------------------------------------------- kernel 4: proj + residual ----
__global__ void proj_kernel(const unsigned short* __restrict__ ob,
                            const unsigned short* __restrict__ wp,
                            const float* __restrict__ bias,
                            const float* __restrict__ x,
                            float* __restrict__ out) {
  __shared__ __align__(16) unsigned short s_w[64 * LDW];   // ~65 KB
  const int lane = threadIdx.x & 31;
  const int wave = threadIdx.x >> 5;
  const int m0 = (blockIdx.x * 8 + wave) * 16;   // token rows
  const int c0 = blockIdx.y * 64;                // output channels
  const int b  = blockIdx.z;

  for (int i = threadIdx.x; i < 64 * (CH / 8); i += 256) {
    int r = i >> 6, ck = (i & 63) * 8;
    *(v8bf*)(s_w + r * LDW + ck) =
        *(const v8bf*)((const bf16*)wp + (size_t)(c0 + r) * CH + ck);
  }
  __syncthreads();

  const unsigned short* A = ob + ((size_t)b * NTOK + m0) * CH;
  v8f acc[4] = {v8f{0}, v8f{0}, v8f{0}, v8f{0}};
  v16bf a_cur = load_a_frag(A, CH, lane);
  for (int k = 0; k < CH; k += 32) {
    v16bf a_nxt = load_a_frag(A + ((k + 32) & (CH - 1)), CH, lane);
    v16bf bfr[4];
#pragma unroll
    for (int t = 0; t < 4; ++t)
      bfr[t] = load_b_frag16(s_w + (size_t)(t * 16) * LDW + k, LDW, lane);
#pragma unroll
    for (int t = 0; t < 4; ++t) acc[t] = wmma_bf16(a_cur, bfr[t], acc[t]);
    a_cur = a_nxt;
  }
  const int col = lane & 15, hi = lane >> 4;
#pragma unroll
  for (int t = 0; t < 4; ++t) {
    int c = c0 + t * 16 + col;
    float bv = bias[c];
    const float* xr = x + ((size_t)b * CH + c) * NTOK;
    float* orow = out + ((size_t)b * CH + c) * NTOK;
#pragma unroll
    for (int v = 0; v < 8; ++v) {
      int n = m0 + v + 8 * hi;
      orow[n] = xr[n] + acc[t][v] + bv;
    }
  }
}

// ---------------------------------------------------------------- launch ----
extern "C" void kernel_launch(void* const* d_in, const int* in_sizes, int n_in,
                              void* d_out, int out_size, void* d_ws,
                              size_t ws_size, hipStream_t stream) {
  (void)in_sizes; (void)n_in; (void)out_size; (void)ws_size;
  const float* x      = (const float*)d_in[0];
  const float* norm_w = (const float*)d_in[1];
  const float* w_qkv  = (const float*)d_in[2];
  const float* b_qkv  = (const float*)d_in[3];
  const float* w_proj = (const float*)d_in[4];
  const float* b_proj = (const float*)d_in[5];
  float* out = (float*)d_out;

  char* ws = (char*)d_ws;
  const size_t SZ_ACT = (size_t)BATCH * NTOK * CH * 2;   // 8 MB each
  unsigned short* xn = (unsigned short*)(ws);
  unsigned short* qb = (unsigned short*)(ws + 1 * SZ_ACT);
  unsigned short* kb = (unsigned short*)(ws + 2 * SZ_ACT);
  unsigned short* vt = (unsigned short*)(ws + 3 * SZ_ACT);
  unsigned short* ob = (unsigned short*)(ws + 4 * SZ_ACT);
  unsigned short* wq = (unsigned short*)(ws + 5 * SZ_ACT);
  unsigned short* wp = (unsigned short*)(ws + 5 * SZ_ACT + (size_t)DQKV * CH * 2);

  cvt_f32_bf16<<<(DQKV * CH + 255) / 256, 256, 0, stream>>>(w_qkv, wq,
                                                            DQKV * CH);
  cvt_f32_bf16<<<(CH * CH + 255) / 256, 256, 0, stream>>>(w_proj, wp, CH * CH);
  ln_kernel<<<(BATCH * NTOK) / 64, 256, 0, stream>>>(x, norm_w, xn);
  qkv_gemm<<<dim3(NTOK / 128, DQKV / 64, BATCH), 256, 0, stream>>>(
      xn, wq, b_qkv, qb, kb, vt);
  attn_kernel<<<dim3(NTOK / 32, BATCH), 256, 0, stream>>>(qb, kb, vt, ob);
  proj_kernel<<<dim3(NTOK / 128, CH / 64, BATCH), 256, 0, stream>>>(
      ob, wp, b_proj, x, out);
}
